// CasualSelfAttention_63780264346015
// MI455X (gfx1250) — compile-verified, matrix operand order checked by reading, not run on hardware
//
#include <hip/hip_runtime.h>
#include <hip/hip_bf16.h>

// ---------------------------------------------------------------------------
// CDNA5 (gfx1250) fused attention block:
//   0) one-shot fp32 -> bf16 conversion of x / w_attn / w_proj
//   1) qkv = x @ w_attn + b_attn     (bf16 WMMA, async-to-LDS A tiles,
//                                     scatter to [3][B,H,T,d] bf16)
//   2) flash attention per (b,h,16-row q tile): S via WMMA, online softmax,
//      V tiles staged with global_load_async_to_lds_b128, O += P*V via WMMA
//   3) out = attn @ w_proj + b_proj  (bf16 WMMA, fp32 output)
// B=8 T=1024 C=768 H=12 d=64
// ---------------------------------------------------------------------------

typedef __attribute__((ext_vector_type(16))) __bf16 v16bf;
typedef __attribute__((ext_vector_type(8)))  float  v8f;

union FragAB { v16bf v; unsigned int u[8]; __bf16 h[16]; };
union FragC  { v8f   v; float f[8]; };

__device__ __forceinline__ v8f wmma_bf16(v16bf a, v16bf b, v8f c) {
  // D = A(16x32 bf16) * B(32x16 bf16) + C(16x16 f32)
  return __builtin_amdgcn_wmma_f32_16x16x32_bf16(
      /*neg_a=*/false, a, /*neg_b=*/false, b,
      /*c_mod=*/(short)0, c, /*reuse_a=*/false, /*reuse_b=*/false);
}

// Async global->LDS copy, 16 bytes per lane. Tracked by ASYNCcnt.
__device__ __forceinline__ void async_ld_b128(unsigned lds_off, const void* g) {
  asm volatile("global_load_async_to_lds_b128 %0, %1, off"
               :: "v"(lds_off), "v"(g) : "memory");
}
__device__ __forceinline__ void wait_async0() {
  asm volatile("s_wait_asynccnt 0" ::: "memory");
}

// Per-lane K-pair base for 16-bit A/B fragment layout (ISA 7.12.2).
__device__ __forceinline__ int kpair(int j, int g) {
  return (j < 4) ? (2 * j + 8 * g) : (16 + 2 * (j - 4) + 8 * g);
}

__device__ __forceinline__ float rmax16(float v) {
  v = fmaxf(v, __shfl_xor(v, 1, 32));
  v = fmaxf(v, __shfl_xor(v, 2, 32));
  v = fmaxf(v, __shfl_xor(v, 4, 32));
  v = fmaxf(v, __shfl_xor(v, 8, 32));
  return v;
}
__device__ __forceinline__ float rsum16(float v) {
  v += __shfl_xor(v, 1, 32);
  v += __shfl_xor(v, 2, 32);
  v += __shfl_xor(v, 4, 32);
  v += __shfl_xor(v, 8, 32);
  return v;
}

#define BB 8
#define TT 1024
#define CC 768
#define HH 12
#define DD 64
#define S3 ((size_t)BB * HH * TT * DD)  // elements per q/k/v plane

// ---------------------------------------------------------------------------
// fp32 -> bf16 elementwise (4 elems / thread); n4 = n/4
// ---------------------------------------------------------------------------
__global__ __launch_bounds__(256) void cvt_bf16(const float* __restrict__ s,
                                                __bf16* __restrict__ d, int n4) {
  const int i = blockIdx.x * 256 + threadIdx.x;
  if (i < n4) {
    const float4 v = ((const float4*)s)[i];
    union { __bf16 h[4]; uint2 u; } t;
    t.h[0] = (__bf16)v.x; t.h[1] = (__bf16)v.y;
    t.h[2] = (__bf16)v.z; t.h[3] = (__bf16)v.w;
    ((uint2*)d)[i] = t.u;
  }
}

// ---------------------------------------------------------------------------
// Tiled GEMM:  D = A[MxK](bf16) * B[KxN](bf16) + bias[N](f32)
// MODE 0: epilogue scatters bf16 into qkv [3][B,H,T,d]
// MODE 1: epilogue writes fp32 row-major [MxN]
// Block tile 128x128, K-step 32, 256 threads = 8 waves, wave tile 32x64.
// A tile fetched with global_load_async_to_lds_b128 (no VGPR round-trip).
// ---------------------------------------------------------------------------
template <int MODE>
__global__ __launch_bounds__(256) void gemm_wmma(const __bf16* __restrict__ Aptr,
                                                 const __bf16* __restrict__ Bptr,
                                                 const float* __restrict__ bias,
                                                 void* __restrict__ Dptr,
                                                 int M, int N, int K) {
  __shared__ __bf16 As[128][32];   // [m][k]
  __shared__ __bf16 Bs[128][32];   // [n][k]  (B tile stored transposed)

  const int tid  = threadIdx.x;
  const int bm   = blockIdx.x * 128;
  const int bn   = blockIdx.y * 128;
  const int w    = tid >> 5;
  const int lane = tid & 31;
  const int wm   = (w & 3) * 32;   // 4 waves along M
  const int wn   = (w >> 2) * 64;  // 2 waves along N
  const int lrow = lane & 15;
  const int g    = lane >> 4;

  FragC acc[2][4];
#pragma unroll
  for (int mi = 0; mi < 2; ++mi)
#pragma unroll
    for (int ni = 0; ni < 4; ++ni)
#pragma unroll
      for (int r = 0; r < 8; ++r) acc[mi][ni].f[r] = 0.0f;

  for (int kb = 0; kb < K; kb += 32) {
    // --- A tile: async global->LDS, 2 x b128 per thread (8 KB total) ---
    {
      const int m    = tid >> 1;
      const int half = tid & 1;
      const __bf16* ga = Aptr + (size_t)(bm + m) * K + kb + half * 16;
      const unsigned lo = (unsigned)(size_t)&As[m][half * 16];
      async_ld_b128(lo, ga);
      async_ld_b128(lo + 16, ga + 8);
    }
    // --- B tile: regular loads, transposed store [n][k] ---
    {
      const int k  = tid >> 3;
      const int ns = (tid & 7) * 16;
      const __bf16* bp = Bptr + (size_t)(kb + k) * N + bn + ns;
#pragma unroll
      for (int i = 0; i < 16; ++i) Bs[ns + i][k] = bp[i];
    }
    wait_async0();
    __syncthreads();

    // --- fragments per documented CDNA5 16-bit A/B layouts ---
    FragAB afr[2], bfr[4];
#pragma unroll
    for (int mi = 0; mi < 2; ++mi) {
      const int row = wm + mi * 16 + lrow;
#pragma unroll
      for (int j = 0; j < 8; ++j)
        afr[mi].u[j] = *(const unsigned int*)&As[row][kpair(j, g)];
    }
#pragma unroll
    for (int ni = 0; ni < 4; ++ni) {
      const int col = wn + ni * 16 + lrow;
#pragma unroll
      for (int j = 0; j < 8; ++j)
        bfr[ni].u[j] = *(const unsigned int*)&Bs[col][kpair(j, g)];
    }
#pragma unroll
    for (int mi = 0; mi < 2; ++mi)
#pragma unroll
      for (int ni = 0; ni < 4; ++ni)
        acc[mi][ni].v = wmma_bf16(afr[mi].v, bfr[ni].v, acc[mi][ni].v);
    __syncthreads();
  }

  // --- epilogue ---
#pragma unroll
  for (int mi = 0; mi < 2; ++mi)
#pragma unroll
    for (int ni = 0; ni < 4; ++ni)
#pragma unroll
      for (int r = 0; r < 8; ++r) {
        const int mg = bm + wm + mi * 16 + r + 8 * g;
        const int ng = bn + wn + ni * 16 + lrow;
        const float val = acc[mi][ni].f[r] + bias[ng];
        if (MODE == 0) {
          const int sel = ng / CC;           // 0=q 1=k 2=v
          const int c   = ng - sel * CC;
          const int h   = c >> 6;
          const int dd  = c & 63;
          const int b   = mg >> 10;          // mg = b*T + t
          const int t   = mg & 1023;
          __bf16* q = (__bf16*)Dptr;
          q[(size_t)sel * S3 + (((size_t)(b * HH + h) * TT + t) << 6) + dd] =
              (__bf16)val;
        } else {
          ((float*)Dptr)[(size_t)mg * N + ng] = val;
        }
      }
}

// ---------------------------------------------------------------------------
// Fused flash attention. One wave owns 16 query rows of one (b,h).
// Key loop step = 32: V tile async-staged to LDS at loop top (overlapped with
// S = Q*K^T WMMA + online softmax), P re-layout C->A via per-wave LDS
// (s_wait_dscnt), O += P*V via WMMA from the staged V tile.
// Reference-faithful mask: score==0 -> -inf (i.e. NO causal mask otherwise).
// ---------------------------------------------------------------------------
__global__ __launch_bounds__(256) void flash_attn(const __bf16* __restrict__ qkv,
                                                  __bf16* __restrict__ attn_out) {
  __shared__ __bf16 pstage[8][16 * 32];  // per-wave P staging   (1 KB each)
  __shared__ __bf16 vstage[8][32 * 64];  // per-wave V tile      (4 KB each)

  const int tid  = threadIdx.x;
  const int w    = tid >> 5;
  const int lane = tid & 31;
  const int gw   = blockIdx.x * 8 + w;   // 0 .. 6143
  const int qt   = gw & 63;              // 64 query tiles of 16
  const int bh   = gw >> 6;              // 0 .. 95
  const int lrow = lane & 15;
  const int g    = lane >> 4;

  const __bf16* qp = qkv + ((size_t)bh * TT + qt * 16) * DD;
  const __bf16* kp = qkv + S3 + (size_t)bh * TT * DD;
  const __bf16* vp = qkv + 2 * S3 + (size_t)bh * TT * DD;
  __bf16* pb = pstage[w];
  __bf16* vw = vstage[w];
  const unsigned vrow_lds = (unsigned)(size_t)(vw + lane * DD);  // this lane's row

  // Q fragments: 16x64 -> two 16x32 A-fragments, held for the whole pass.
  FragAB qa[2];
#pragma unroll
  for (int kk = 0; kk < 2; ++kk)
#pragma unroll
    for (int j = 0; j < 8; ++j)
      qa[kk].u[j] = *(const unsigned int*)(qp + (size_t)lrow * DD + kk * 32 + kpair(j, g));

  const float scale = 0.125f;      // d^-0.5, d=64
  const float NEG   = -3.0e38f;    // "-inf" that keeps exp/alpha finite

  float mrun[8], lrun[8];
  FragC o[4];
#pragma unroll
  for (int r = 0; r < 8; ++r) { mrun[r] = NEG; lrun[r] = 0.0f; }
#pragma unroll
  for (int ni = 0; ni < 4; ++ni)
#pragma unroll
    for (int r = 0; r < 8; ++r) o[ni].f[r] = 0.0f;

  for (int jb = 0; jb < TT; jb += 32) {
    // ---- kick off async V tile copy: lane owns key-row (jb+lane), 128 B ----
    {
      const __bf16* gv = vp + (size_t)(jb + lane) * DD;
#pragma unroll
      for (int c = 0; c < 8; ++c)
        async_ld_b128(vrow_lds + c * 16, gv + c * 8);
    }

    // ---- S = Q * K^T for 32 keys: two 16x16 C tiles (overlaps async V) ----
    FragC s0, s1;
#pragma unroll
    for (int r = 0; r < 8; ++r) { s0.f[r] = 0.0f; s1.f[r] = 0.0f; }
    FragAB kf;
#pragma unroll
    for (int half = 0; half < 2; ++half) {
#pragma unroll
      for (int kk = 0; kk < 2; ++kk) {
        const size_t keyrow = (size_t)(jb + half * 16 + lrow) * DD;
#pragma unroll
        for (int j = 0; j < 8; ++j)
          kf.u[j] = *(const unsigned int*)(kp + keyrow + kk * 32 + kpair(j, g));
        if (half == 0) s0.v = wmma_bf16(qa[kk].v, kf.v, s0.v);
        else           s1.v = wmma_bf16(qa[kk].v, kf.v, s1.v);
      }
    }

    // ---- online softmax (rows r+8g, cols = lane group of 16) ----
    float p0[8], p1[8], alpha[8];
#pragma unroll
    for (int r = 0; r < 8; ++r) {
      float a = s0.f[r];
      float b = s1.f[r];
      a = (a == 0.0f) ? NEG : a * scale;   // faithful score==0 mask
      b = (b == 0.0f) ? NEG : b * scale;
      float rm   = rmax16(fmaxf(a, b));
      float mnew = fmaxf(mrun[r], rm);
      alpha[r]   = __expf(mrun[r] - mnew);
      p0[r]      = __expf(a - mnew);
      p1[r]      = __expf(b - mnew);
      const float rs = rsum16(p0[r] + p1[r]);
      lrun[r] = lrun[r] * alpha[r] + rs;
      mrun[r] = mnew;
    }

    // ---- P: C-layout -> A-layout via per-wave LDS ----
#pragma unroll
    for (int r = 0; r < 8; ++r) {
      const int m = r + 8 * g;
      pb[m * 32 + lrow]      = (__bf16)p0[r];
      pb[m * 32 + 16 + lrow] = (__bf16)p1[r];
    }
    asm volatile("s_wait_dscnt 0" ::: "memory");  // cross-lane LDS RAW
    FragAB pa;
#pragma unroll
    for (int j = 0; j < 8; ++j)
      pa.u[j] = *(const unsigned int*)(pb + lrow * 32 + kpair(j, g));
    asm volatile("" ::: "memory");                // keep reads before next stores

    // ---- V tile ready? then O = O*alpha + P*V ----
    wait_async0();
#pragma unroll
    for (int ni = 0; ni < 4; ++ni) {
#pragma unroll
      for (int r = 0; r < 8; ++r) o[ni].f[r] *= alpha[r];
      FragAB vf;
#pragma unroll
      for (int j = 0; j < 8; ++j) {
        const int k0 = kpair(j, g);
        vf.h[2 * j]     = vw[k0 * DD + ni * 16 + lrow];
        vf.h[2 * j + 1] = vw[(k0 + 1) * DD + ni * 16 + lrow];
      }
      o[ni].v = wmma_bf16(pa.v, vf.v, o[ni].v);
    }
  }

  // ---- normalize and store bf16 [B*T, C] with channel = h*64 + d ----
  const int b = bh / HH;
  const int h = bh - b * HH;
#pragma unroll
  for (int r = 0; r < 8; ++r) {
    const float inv = 1.0f / lrun[r];
    const int   t   = qt * 16 + r + 8 * g;
    const size_t row = (size_t)(b * TT + t) * CC;
#pragma unroll
    for (int ni = 0; ni < 4; ++ni)
      attn_out[row + h * DD + ni * 16 + lrow] = (__bf16)(o[ni].f[r] * inv);
  }
}

// ---------------------------------------------------------------------------
extern "C" void kernel_launch(void* const* d_in, const int* in_sizes, int n_in,
                              void* d_out, int out_size, void* d_ws, size_t ws_size,
                              hipStream_t stream) {
  (void)in_sizes; (void)n_in; (void)out_size; (void)ws_size;
  const float* x      = (const float*)d_in[0];
  const float* w_attn = (const float*)d_in[1];
  const float* b_attn = (const float*)d_in[2];
  const float* w_proj = (const float*)d_in[3];
  const float* b_proj = (const float*)d_in[4];
  float* out = (float*)d_out;

  // workspace layout (bf16 elements)
  __bf16* qkv      = (__bf16*)d_ws;                 // 3*S3          (~37.7 MB)
  __bf16* attn_out = qkv + 3 * S3;                  // B*T*C         (~12.6 MB)
  __bf16* x_bf     = attn_out + (size_t)BB * TT * CC;        // B*T*C
  __bf16* wa_bf    = x_bf + (size_t)BB * TT * CC;            // C*3C
  __bf16* wp_bf    = wa_bf + (size_t)CC * 3 * CC;            // C*C

  const int n_x  = BB * TT * CC;      // 6291456
  const int n_wa = CC * 3 * CC;       // 1769472
  const int n_wp = CC * CC;           // 589824

  // 0) one-shot fp32 -> bf16 conversions
  cvt_bf16<<<(n_x / 4 + 255) / 256, 256, 0, stream>>>(x, x_bf, n_x / 4);
  cvt_bf16<<<(n_wa / 4 + 255) / 256, 256, 0, stream>>>(w_attn, wa_bf, n_wa / 4);
  cvt_bf16<<<(n_wp / 4 + 255) / 256, 256, 0, stream>>>(w_proj, wp_bf, n_wp / 4);

  // 1) qkv = x @ w_attn + b_attn   (M=8192, N=2304, K=768)
  dim3 g1(BB * TT / 128, 3 * CC / 128);
  gemm_wmma<0><<<g1, 256, 0, stream>>>(x_bf, wa_bf, b_attn, qkv,
                                       BB * TT, 3 * CC, CC);

  // 2) fused flash attention: 96 (b,h) * 64 q-tiles = 6144 waves / 8 per block
  flash_attn<<<BB * HH * (TT / 16) / 8, 256, 0, stream>>>(qkv, attn_out);

  // 3) out = attn @ w_proj + b_proj  (M=8192, N=768, K=768), fp32 out
  dim3 g3(BB * TT / 128, CC / 128);
  gemm_wmma<1><<<g3, 256, 0, stream>>>(attn_out, wp_bf, b_proj, out,
                                       BB * TT, CC, CC);
}
